// GNNConvEBM_15857019257193
// MI455X (gfx1250) — matrix operand order repaired and checked by reading, not compiled, not run on hardware
//
#include <hip/hip_runtime.h>
#include <hip/hip_bf16.h>
#include <math.h>

typedef _Float16 f16;
typedef __attribute__((ext_vector_type(8)))  _Float16 v8h;
typedef __attribute__((ext_vector_type(16))) _Float16 v16h;
typedef __attribute__((ext_vector_type(8)))  float    v8f;

#define OD 32
#define NN 48
#define TT 16
#define BB 4
#define BT (BB*TT)

// sigmoid via HW transcendental + HW reciprocal approximation:
// one v_exp_f32 + one v_rcp_f32, no IEEE div expansion.
__device__ __forceinline__ float sigf(float x) {
  return __builtin_amdgcn_rcpf(1.0f + __expf(-x));
}

// D = A(16x32 f16) * B(32x16 f16) + C(f32), wave32 WMMA
__device__ __forceinline__ v8f wmma_f16(v16h a, v16h b, v8f c) {
  return __builtin_amdgcn_wmma_f32_16x16x32_f16(
      /*neg_a=*/false, a, /*neg_b=*/false, b,
      /*c_mod=*/(short)0, c, /*reuse_a=*/false, /*reuse_b=*/false);
}

// A fragment: 16x32 f16 tile at base (row stride ld halves, 16B-aligned rows).
// ISA 7.12.2: lane l<16 -> M=l, K={0..7}u{16..23}; lane l>=16 -> M=l-16, K={8..15}u{24..31}
// Two contiguous 8-half runs per lane -> two ds_load_b128, no repacking.
__device__ __forceinline__ v16h load_A(const f16* base, int ld, int lane) {
  int m  = lane & 15;
  int kq = (lane >> 4) ? 8 : 0;
  const f16* r = base + m * ld + kq;
  v8h lo = *(const v8h*)r;          // K = kq .. kq+7
  v8h hi = *(const v8h*)(r + 16);   // K = 16+kq .. 16+kq+7
  return __builtin_shufflevector(lo, hi, 0, 1, 2, 3, 4, 5, 6, 7,
                                 8, 9, 10, 11, 12, 13, 14, 15);
}

// B fragments are pre-swizzled in LDS: tile t stores lane l's 16 halves
// contiguously -> single aligned 32B vector load per lane.
__device__ __forceinline__ v16h load_Bsz(const v16h* buf, int tile, int lane) {
  return buf[tile * 32 + lane];
}

// Stage a KxOD f32 weight matrix into swizzled B-fragment order (f16),
// zero-padding rows k >= kvalid. Tiles ordered (ks, nt), ks-major.
// Fragment layout: lane l (0-15) -> N=nt*16+l, K=ks*32+{0..7,16..23}[j];
//                  lane l (16-31) -> N=nt*16+(l-16), K=ks*32+{8..15,24..31}[j].
__device__ __forceinline__ void stage_B(v16h* buf, const float* __restrict__ W,
                                        int ktiles, int kvalid, int tid,
                                        int nthreads) {
  f16* h = (f16*)buf;
  int total = ktiles * 2 * 512;
  for (int idx = tid; idx < total; idx += nthreads) {
    int tile = idx >> 9;
    int r    = idx & 511;
    int lane = r >> 4, j = r & 15;
    int ks = tile >> 1, nt = tile & 1;
    int k = ks * 32 + ((lane >> 4) << 3) + (j & 7) + ((j >> 3) << 4);
    int n = nt * 16 + (lane & 15);
    float v = (k < kvalid) ? W[k * OD + n] : 0.0f;
    h[idx] = (f16)v;
  }
}

// ---------------------------------------------------------------- temb ------
__global__ __launch_bounds__(32) void temb_kernel(
    const float* __restrict__ t,
    const float* __restrict__ tw1, const float* __restrict__ tb1,
    const float* __restrict__ tw2, const float* __restrict__ tb2,
    float* __restrict__ temb) {
  __shared__ float h1[OD];
  int b = blockIdx.x;
  int c = threadIdx.x;
  float tv = t[b];
  const float coef = -logf(10000.0f) / 255.0f;  // half-1 = 255
  float acc = tb1[c];
  for (int k = 0; k < 512; ++k) {
    int j   = (k < 256) ? k : (k - 256);
    float f = __expf((float)j * coef);
    float e = tv * f;
    float s = (k < 256) ? __sinf(e) : __cosf(e);
    acc += s * tw1[k * OD + c];
  }
  float g = 0.5f * acc * (1.0f + erff(acc * 0.70710678f));  // exact gelu
  h1[c] = g;
  __syncthreads();
  float acc2 = tb2[c];
#pragma unroll
  for (int d = 0; d < OD; ++d) acc2 += h1[d] * tw2[d * OD + c];
  temb[b * OD + c] = acc2;
}

// ------------------------------------------------------------- assemble -----
__global__ __launch_bounds__(64) void assemble_kernel(
    const float* __restrict__ inp,   // (B,N,16)
    const float* __restrict__ outp,  // (B,T,N,16)
    const float* __restrict__ temb,  // (B,32)
    float* __restrict__ x) {         // (B,T,N,64)
  int r = blockIdx.x;  // flat (b,t,n)
  int c = threadIdx.x;
  int n = r % NN;
  int bt = r / NN;
  int t = bt % TT;
  int b = bt / TT;
  float v;
  if (c < 16)      v = inp[(b * NN + n) * 16 + c];
  else if (c < 32) v = outp[((b * TT + t) * NN + n) * 16 + (c - 16)];
  else             v = temb[b * OD + (c - 32)];
  x[r * 64 + c] = v;
}

// ------------------------------------------------------------------ NLM -----
// One workgroup (256 threads = 8 wave32) per (b,t) slice.
template <int H>
__global__ __launch_bounds__(256) void nlm_kernel(
    const float* __restrict__ xin,  // (BT, 48, H)
    const float* __restrict__ w10,  // (2H, 32)
    const float* __restrict__ b10,
    const float* __restrict__ w11,  // (H+6, 32)
    const float* __restrict__ b11,
    const float* __restrict__ w12,  // (2H+6, 32)
    const float* __restrict__ b12,
    const float* __restrict__ w21,  // (128, 32)
    const float* __restrict__ b21,
    float* __restrict__ yout) {     // (BT, 48, 32)
  constexpr int K1  = H + 32;   // padded K for the x1 GEMM (cols H..H+5 = reduce2(cmp))
  constexpr int KT1 = K1 / 32;  // K-tiles for x1
  constexpr int KTH = H / 32;   // K-tiles for P/Q
  __shared__ __align__(32) f16 xh[NN][K1];
  __shared__ __align__(32) f16 f1h[NN][128];  // [x0 | x1 | interleaved max/min of x2]
  __shared__ v16h w11sz[KT1 * 2 * 32];
  __shared__ v16h wAsz[KTH * 2 * 32];
  __shared__ v16h wCsz[KTH * 2 * 32];
  __shared__ v16h w21sz[4 * 2 * 32];
  __shared__ float Pm[NN][OD];
  __shared__ float Qm[NN][OD];
  __shared__ float nums[NN];
  __shared__ float r1buf[2 * H];
  __shared__ float b11s[OD], b12s[OD], b21s[OD];
  __shared__ float cB0[OD], cB2[OD], cD0[OD], cD2[OD];

  const int bt   = blockIdx.x;
  const int tid  = threadIdx.x;
  const int lane = tid & 31;
  const int wave = tid >> 5;
  const float* xb = xin + (size_t)bt * NN * H;

  // ---- Phase 1: stage x (f32->f16) and weights into LDS -------------------
  for (int idx = tid; idx < NN * K1; idx += 256) {
    int i = idx / K1, k = idx - i * K1;
    float v = (k < H) ? xb[i * H + k] : 0.0f;
    __builtin_prefetch(xb + i * H + k + 256, 0, 1);
    xh[i][k] = (f16)v;
    if (k == 1) nums[i] = v;
  }
  stage_B(w11sz, w11, KT1, H + 6, tid, 256);
  stage_B(wAsz, w12, KTH, H, tid, 256);                 // rows 0..H-1
  stage_B(wCsz, w12 + (H + 3) * OD, KTH, H, tid, 256);  // rows H+3..2H+2
  stage_B(w21sz, w21, 4, 128, tid, 256);
  if (tid < OD) {
    b11s[tid] = b11[tid];
    b12s[tid] = b12[tid];
    b21s[tid] = b21[tid];
    cB0[tid] = w12[(H + 0) * OD + tid];      // cmp_ij channel (i<j)
    cB2[tid] = w12[(H + 2) * OD + tid];      // cmp_ij channel (nums_i<nums_j)
    cD0[tid] = w12[(2 * H + 3) * OD + tid];  // cmp_ji channel (j<i)
    cD2[tid] = w12[(2 * H + 5) * OD + tid];  // cmp_ji channel (nums_j<nums_i)
  }
  __syncthreads();

  // ---- Phase 2: reduce2(cmp) into xh cols H..H+5; reduce1 into r1buf ------
  if (tid < NN) {
    int i = tid;
    float ni = nums[i];
    float mx0 = -1.0f, mn0 = 2.0f, mx2 = -1.0f, mn2 = 2.0f;
    for (int j = 0; j < NN; ++j) {
      if (j == i) continue;
      float c0 = (i < j) ? 1.0f : 0.0f;
      float c2 = (ni < nums[j]) ? 1.0f : 0.0f;
      mx0 = fmaxf(mx0, c0); mn0 = fminf(mn0, c0);
      mx2 = fmaxf(mx2, c2); mn2 = fminf(mn2, c2);
    }
    xh[i][H + 0] = (f16)mx0; xh[i][H + 1] = (f16)mn0;
    xh[i][H + 2] = (f16)0.0f; xh[i][H + 3] = (f16)0.0f;  // (i==j) masked out
    xh[i][H + 4] = (f16)mx2; xh[i][H + 5] = (f16)mn2;
  }
  if (tid < H) {
    int k = tid;
    float mx = -1e30f, mn = 1e30f;
    for (int i = 0; i < NN; ++i) {
      float v = (float)xh[i][k];
      mx = fmaxf(mx, v);
      mn = fminf(mn, v);
    }
    r1buf[2 * k] = mx;
    r1buf[2 * k + 1] = mn;
  }
  __syncthreads();

  // ---- Phase 3: WMMA — x1 = sigma(x|r2cmp @ w11), P = x@wA, Q = x@wC ------
  // 18 (Mtile,Ntile) tasks round-robined over 8 waves.
  for (int task = wave; task < 18; task += 8) {
    int kind = task / 6;  // 0=x1, 1=P, 2=Q
    int tt   = task % 6;
    int mt = tt >> 1, nt = tt & 1;
    v8f acc = {};
    if (kind == 0) {
#pragma unroll
      for (int ks = 0; ks < KT1; ++ks) {
        v16h a = load_A(&xh[mt * 16][ks * 32], K1, lane);
        v16h b = load_Bsz(w11sz, ks * 2 + nt, lane);
        acc = wmma_f16(a, b, acc);
      }
      int n  = nt * 16 + (lane & 15);
      int m0 = mt * 16 + ((lane >> 4) ? 8 : 0);
#pragma unroll
      for (int v = 0; v < 8; ++v)
        f1h[m0 + v][OD + n] = (f16)sigf(acc[v] + b11s[n]);
    } else {
      const v16h* W = (kind == 1) ? wAsz : wCsz;
#pragma unroll
      for (int ks = 0; ks < KTH; ++ks) {
        v16h a = load_A(&xh[mt * 16][ks * 32], K1, lane);
        v16h b = load_Bsz(W, ks * 2 + nt, lane);
        acc = wmma_f16(a, b, acc);
      }
      float (*Dst)[OD] = (kind == 1) ? Pm : Qm;
      int n  = nt * 16 + (lane & 15);
      int m0 = mt * 16 + ((lane >> 4) ? 8 : 0);
#pragma unroll
      for (int v = 0; v < 8; ++v) Dst[m0 + v][n] = acc[v];
    }
  }
  __syncthreads();

  // ---- Phase 4: x0 = sigma(reduce1(x) @ w10 + b10), broadcast to all rows -
  if (tid < OD) {
    int c = tid;
    float acc = b10[c];
    for (int d = 0; d < 2 * H; ++d) acc += r1buf[d] * w10[d * OD + c];
    f16 v = (f16)sigf(acc);
    for (int i = 0; i < NN; ++i) f1h[i][c] = v;
  }
  // ---- Phase 5: reduce2(x2) via sigmoid-monotonicity ----------------------
  // x2[i,j,c] = sigma(P[i,c] + Q[j,c] + C(i,j,c) + b12[c]); max/min commute with sigma.
  for (int p = tid; p < NN * OD; p += 256) {
    int i = p >> 5, c = p & 31;
    float pic = Pm[i][c] + b12s[c];
    float qb0 = cB0[c], qb2 = cB2[c], qd0 = cD0[c], qd2 = cD2[c];
    float ni = nums[i];
    float mx = -1e30f, mn = 1e30f;
    for (int j = 0; j < NN; ++j) {
      if (j == i) continue;
      float nj = nums[j];
      float v = Qm[j][c] + ((i < j) ? qb0 : qd0) + ((ni < nj) ? qb2 : 0.0f) +
                ((nj < ni) ? qd2 : 0.0f);
      mx = fmaxf(mx, v);
      mn = fminf(mn, v);
    }
    f1h[i][64 + 2 * c]     = (f16)sigf(pic + mx);
    f1h[i][64 + 2 * c + 1] = (f16)sigf(pic + mn);
  }
  __syncthreads();

  // ---- Phase 6: y = swish(sigma(f1 @ w21 + b21)) --------------------------
  for (int task = wave; task < 6; task += 8) {
    int mt = task >> 1, nt = task & 1;
    v8f acc = {};
#pragma unroll
    for (int ks = 0; ks < 4; ++ks) {
      v16h a = load_A(&f1h[mt * 16][ks * 32], 128, lane);
      v16h b = load_Bsz(w21sz, ks * 2 + nt, lane);
      acc = wmma_f16(a, b, acc);
    }
    int n  = nt * 16 + (lane & 15);
    int m0 = mt * 16 + ((lane >> 4) ? 8 : 0);
#pragma unroll
    for (int v = 0; v < 8; ++v) {
      float s = sigf(acc[v] + b21s[n]);
      yout[((size_t)bt * NN + (m0 + v)) * OD + n] = s * sigf(s);
    }
  }
}

// ----------------------------------------------------------------- conv -----
// conv1d over T (kernel 3, pad 1) as a 16x96x32 WMMA GEMM per (b,n).
__global__ __launch_bounds__(64) void conv_kernel(
    const float* __restrict__ yin,  // (B,T,N,32)
    const float* __restrict__ cw,   // (32,32,3) [co][ci][dt]
    const float* __restrict__ cb,   // (32)
    float* __restrict__ xout) {     // (B,T,N,32)
  __shared__ __align__(32) f16 arow[TT][96];  // row t: [y[t-1] | y[t] | y[t+1]]
  __shared__ v16h wksz[3 * 2 * 32];
  __shared__ float cbs[OD];
  int bn = blockIdx.x;
  int b = bn / NN, nidx = bn % NN;
  int tid = threadIdx.x, lane = tid & 31, wave = tid >> 5;

  for (int idx = tid; idx < TT * 96; idx += 64) {
    int t = idx / 96, k = idx - t * 96;
    int dt = k / 32 - 1, ci = k & 31;
    int t2 = t + dt;
    float v = (t2 >= 0 && t2 < TT)
                  ? yin[((size_t)(b * TT + t2) * NN + nidx) * OD + ci]
                  : 0.0f;
    arow[t][k] = (f16)v;
  }
  // Stage conv weight in swizzled B order: K index k = dt*32+ci, N index = co.
  {
    f16* h = (f16*)wksz;
    for (int idx = tid; idx < 3 * 2 * 512; idx += 64) {
      int tile = idx >> 9;
      int r    = idx & 511;
      int ln = r >> 4, j = r & 15;
      int ks = tile >> 1, nt = tile & 1;
      int k  = ks * 32 + ((ln >> 4) << 3) + (j & 7) + ((j >> 3) << 4);
      int co = nt * 16 + (ln & 15);
      int dt = k / 32, ci = k & 31;
      h[idx] = (f16)cw[(co * OD + ci) * 3 + dt];
    }
  }
  if (tid < OD) cbs[tid] = cb[tid];
  __syncthreads();

  int nt = wave;  // wave 0 -> co 0..15, wave 1 -> co 16..31
  v8f acc = {};
#pragma unroll
  for (int ks = 0; ks < 3; ++ks) {
    v16h a  = load_A(&arow[0][ks * 32], 96, lane);
    v16h bm = load_Bsz(wksz, ks * 2 + nt, lane);
    acc = wmma_f16(a, bm, acc);
  }
  int co = nt * 16 + (lane & 15);
  int m0 = (lane >> 4) ? 8 : 0;
#pragma unroll
  for (int v = 0; v < 8; ++v) {
    int t = m0 + v;
    float s = acc[v] + cbs[co];
    xout[((size_t)(b * TT + t) * NN + nidx) * OD + co] = s * sigf(s);
  }
}

// ---------------------------------------------------------------- final -----
__global__ __launch_bounds__(256) void final_kernel(
    const float* __restrict__ x,   // (B,T,N,32)
    const float* __restrict__ fw,  // (32)
    const float* __restrict__ fb,  // (1)
    float* __restrict__ out) {     // (B)
  __shared__ float red[256];
  __shared__ float fws[OD];
  int b = blockIdx.x, tid = threadIdx.x;
  if (tid < OD) fws[tid] = fw[tid];
  __syncthreads();
  float fbv = fb[0];
  float acc = 0.0f;
  for (int r = tid; r < TT * NN; r += 256) {
    const float* row = x + ((size_t)b * TT * NN + r) * OD;
    float d = fbv;
#pragma unroll
    for (int c = 0; c < OD; ++c) d += row[c] * fws[c];
    acc += d;
  }
  red[tid] = acc;
  __syncthreads();
  for (int s = 128; s > 0; s >>= 1) {
    if (tid < s) red[tid] += red[tid + s];
    __syncthreads();
  }
  if (tid == 0) out[b] = red[0];
}

// ---------------------------------------------------------------------------
extern "C" void kernel_launch(void* const* d_in, const int* in_sizes, int n_in,
                              void* d_out, int out_size, void* d_ws,
                              size_t ws_size, hipStream_t stream) {
  (void)in_sizes; (void)n_in; (void)out_size; (void)ws_size;
  const float* inp  = (const float*)d_in[0];  // (4,48,16)
  const float* outp = (const float*)d_in[1];  // (4,16,48,16)
  const float* tvec = (const float*)d_in[2];  // (4,)
  const float* tw1  = (const float*)d_in[3];
  const float* tb1  = (const float*)d_in[4];
  const float* tw2  = (const float*)d_in[5];
  const float* tb2  = (const float*)d_in[6];
  // block1
  const float* w10_1 = (const float*)d_in[7];
  const float* b10_1 = (const float*)d_in[8];
  const float* w11_1 = (const float*)d_in[9];
  const float* b11_1 = (const float*)d_in[10];
  const float* w12_1 = (const float*)d_in[11];
  const float* b12_1 = (const float*)d_in[12];
  const float* w21_1 = (const float*)d_in[13];
  const float* b21_1 = (const float*)d_in[14];
  const float* cw1   = (const float*)d_in[15];
  const float* cb1   = (const float*)d_in[16];
  // block2
  const float* w10_2 = (const float*)d_in[17];
  const float* b10_2 = (const float*)d_in[18];
  const float* w11_2 = (const float*)d_in[19];
  const float* b11_2 = (const float*)d_in[20];
  const float* w12_2 = (const float*)d_in[21];
  const float* b12_2 = (const float*)d_in[22];
  const float* w21_2 = (const float*)d_in[23];
  const float* b21_2 = (const float*)d_in[24];
  const float* cw2   = (const float*)d_in[25];
  const float* cb2   = (const float*)d_in[26];
  const float* fw    = (const float*)d_in[27];
  const float* fb    = (const float*)d_in[28];

  float* ws    = (float*)d_ws;
  float* temb  = ws;              // 128
  float* x0buf = temb + 128;      // 4*16*48*64 = 196608
  float* y1    = x0buf + 196608;  // 98304
  float* x2b   = y1 + 98304;      // 98304
  float* y2    = x2b + 98304;     // 98304
  float* x3b   = y2 + 98304;      // 98304

  temb_kernel<<<BB, 32, 0, stream>>>(tvec, tw1, tb1, tw2, tb2, temb);
  assemble_kernel<<<BB * TT * NN, 64, 0, stream>>>(inp, outp, temb, x0buf);
  nlm_kernel<64><<<BT, 256, 0, stream>>>(x0buf, w10_1, b10_1, w11_1, b11_1,
                                         w12_1, b12_1, w21_1, b21_1, y1);
  conv_kernel<<<BB * NN, 64, 0, stream>>>(y1, cw1, cb1, x2b);
  nlm_kernel<32><<<BT, 256, 0, stream>>>(x2b, w10_2, b10_2, w11_2, b11_2,
                                         w12_2, b12_2, w21_2, b21_2, y2);
  conv_kernel<<<BB * NN, 64, 0, stream>>>(y2, cw2, cb2, x3b);
  final_kernel<<<BB, 256, 0, stream>>>(x3b, fw, fb, (float*)d_out);
}
